// CapsLayer_63934883168634
// MI455X (gfx1250) — matrix operand order repaired and checked by reading, not compiled, not run on hardware
//
#include <hip/hip_runtime.h>

// ---------------------------------------------------------------------------
// Capsule layer: vj = squash( sum_i uji ).  Routing softmax is over a size-1
// axis -> coupling coeffs are identically 1.0, so the 3 routing iterations
// are a fixed point and only the I-reduction + squash affect the output.
//
// GEMM: A[64 x 16384] (inputs, contiguous) x Wmat[16384 x 512] -> S[64 x 512]
//   - bf16 WMMA (v_wmma_f32_16x16x32_bf16), fp32 accumulation
//   - one wave per (j, k-slice) computes the full M=64 column block, so the
//     32 MB W matrix streams from HBM exactly once; 4 MB inputs live in L2
//   - split-K = 64; 8 slices share a 256-thread block and reduce partials
//     through LDS, so global fp32 atomics drop 8x (256K total, 8 per addr)
// ---------------------------------------------------------------------------

typedef __attribute__((ext_vector_type(16))) __bf16 v16bf;
typedef __attribute__((ext_vector_type(8)))  float  v8f;
typedef __attribute__((ext_vector_type(4)))  float  v4f;

#define NB 64                    // batch
#define NI 2048                  // input caps
#define NK 8                     // input dim caps
#define NJ 32                    // output caps
#define NL 16                    // output dim caps
#define KTOT (NI * NK)           // 16384 contraction length
#define NCOL (NJ * NL)           // 512 output columns
#define WSTRIDE_I (NJ * NK * NL) // 4096 floats per i in W
#define WSTRIDE_J (NK * NL)      // 128 floats per j in W
#define KSLICES 64               // split-K factor
#define KPER (KTOT / KSLICES)    // 256 K per wave
#define KSTEP 32                 // K per WMMA
#define NSTEPS (KPER / KSTEP)    // 8 WMMA steps (x4 M-tiles) per wave
#define WAVES_PER_BLK 8
#define KGROUPS (KSLICES / WAVES_PER_BLK) // 8

// Pack 16 fp32 (four float4 runs) into a v16bf fragment (v_cvt_pk_bf16_f32).
static __device__ inline v16bf cvt_frag(v4f x0, v4f x1, v4f x2, v4f x3) {
    v16bf r;
    r[0]  = (__bf16)x0[0]; r[1]  = (__bf16)x0[1];
    r[2]  = (__bf16)x0[2]; r[3]  = (__bf16)x0[3];
    r[4]  = (__bf16)x1[0]; r[5]  = (__bf16)x1[1];
    r[6]  = (__bf16)x1[2]; r[7]  = (__bf16)x1[3];
    r[8]  = (__bf16)x2[0]; r[9]  = (__bf16)x2[1];
    r[10] = (__bf16)x2[2]; r[11] = (__bf16)x2[3];
    r[12] = (__bf16)x3[0]; r[13] = (__bf16)x3[1];
    r[14] = (__bf16)x3[2]; r[15] = (__bf16)x3[3];
    return r;
}

__global__ void caps_zero_kernel(float* __restrict__ acc) {
    int t = blockIdx.x * blockDim.x + threadIdx.x;
    if (t < NB * NCOL) acc[t] = 0.0f;
}

// grid: (NJ, KGROUPS), block: 256 (8 waves, one K-slice each)
__global__ __launch_bounds__(256)
void caps_gemm_kernel(const float* __restrict__ in,   // [64, 2048, 8]
                      const float* __restrict__ W,    // [2048, 32, 8, 16]
                      float* __restrict__ acc) {      // [64, 512]
    __shared__ float red[WAVES_PER_BLK * 1024];       // 32 KB of 320 KB LDS

    const int j      = blockIdx.x;                    // 0..31 output capsule
    const int grp    = blockIdx.y;                    // 0..7
    const int tid    = threadIdx.x;
    const int wave   = tid >> 5;
    const int lane   = tid & 31;
    const int laneHi = lane >> 4;
    const int m      = lane & 15;
    const int ks     = grp * WAVES_PER_BLK + wave;    // 0..63 K slice

    v8f accv[4];
#pragma unroll
    for (int t = 0; t < 4; ++t) {
        v8f z = {0.f, 0.f, 0.f, 0.f, 0.f, 0.f, 0.f, 0.f};
        accv[t] = z;
    }

    const int cbase0 = ks * KPER;
    // B base for this lane: Wmat[c, j*16+l] = W[(c>>3)*4096 + j*128 + (c&7)*16 + l]
    // N = lane&15; (c>>3) starts at laneHi and advances by 4 per step.
    const float* pb0 = W + (size_t)((cbase0 >> 3) + laneHi) * WSTRIDE_I
                         + j * WSTRIDE_J + m;

#pragma unroll
    for (int step = 0; step < NSTEPS; ++step) {
        // ---- B fragment: lane&15 = N; e=0..7 -> K=e+laneHi*8, e=8..15 -> K=e+8+laneHi*8.
        const float* pb = pb0 + (size_t)step * 4 * WSTRIDE_I;
        v4f b0 = {pb[0], pb[16], pb[32], pb[48]};
        v4f b1 = {pb[64], pb[80], pb[96], pb[112]};
        v4f b2 = {pb[2 * WSTRIDE_I + 0], pb[2 * WSTRIDE_I + 16], pb[2 * WSTRIDE_I + 32], pb[2 * WSTRIDE_I + 48]};
        v4f b3 = {pb[2 * WSTRIDE_I + 64], pb[2 * WSTRIDE_I + 80], pb[2 * WSTRIDE_I + 96], pb[2 * WSTRIDE_I + 112]};

        // ---- A fragments, one private buffer per M-tile (no reg reuse ->
        //      WMMA hazard slots fill with independent cvts, not v_nops).
        //      16-bit A 16x32 layout: e=0..7 -> K=e+laneHi*8, e=8..15 -> K=e+8+laneHi*8
        v16bf afrag[4];
#pragma unroll
        for (int t = 0; t < 4; ++t) {
            const float* pa = in + (size_t)(t * 16 + m) * KTOT
                                 + cbase0 + step * KSTEP + laneHi * 8;
            v4f a0 = *(const v4f*)(pa + 0);
            v4f a1 = *(const v4f*)(pa + 4);
            v4f a2 = *(const v4f*)(pa + 16);
            v4f a3 = *(const v4f*)(pa + 20);
            afrag[t] = cvt_frag(a0, a1, a2, a3);
        }
        v16bf bfrag = cvt_frag(b0, b1, b2, b3);

#pragma unroll
        for (int t = 0; t < 4; ++t) {
            accv[t] = __builtin_amdgcn_wmma_f32_16x16x32_bf16(
                /*neg_a=*/false, afrag[t], /*neg_b=*/false, bfrag,
                /*c_mod=*/(short)0, accv[t],
                /*reuse_a=*/false, /*reuse_b=*/false);
        }
    }

    // ---- Stage partial tiles in LDS: idx = t*256 + r*32 + lane per wave.
#pragma unroll
    for (int t = 0; t < 4; ++t)
#pragma unroll
        for (int r = 0; r < 8; ++r)
            red[wave * 1024 + t * 256 + r * 32 + lane] = accv[t][r];
    __syncthreads();

    // ---- Cross-wave reduce (8 partials) + one global atomic per output.
    //      C/D layout: lane&15 = N (=l), VGPR r -> M = r + laneHi*8.
#pragma unroll
    for (int i = 0; i < 4; ++i) {
        const int o = tid * 4 + i;                    // 0..1023
        float s = 0.0f;
#pragma unroll
        for (int w = 0; w < WAVES_PER_BLK; ++w) s += red[w * 1024 + o];
        const int t  = o >> 8;
        const int r  = (o >> 5) & 7;
        const int ln = o & 31;
        const int brow = t * 16 + r + ((ln >> 4) << 3);
        const int col  = j * NL + (ln & 15);
        atomicAdd(&acc[(size_t)brow * NCOL + col], s);
    }
}

// One thread per (b, j): squash over the 16-wide L axis.
__global__ void caps_squash_kernel(const float* __restrict__ acc,
                                   float* __restrict__ out) {
    int t = blockIdx.x * blockDim.x + threadIdx.x;    // t = b*NJ + j
    if (t >= NB * NJ) return;
    const float* s = acc + (size_t)t * NL;
    v4f v0 = *(const v4f*)(s + 0);
    v4f v1 = *(const v4f*)(s + 4);
    v4f v2 = *(const v4f*)(s + 8);
    v4f v3 = *(const v4f*)(s + 12);
    float s2 = 0.0f;
#pragma unroll
    for (int l = 0; l < 4; ++l)
        s2 += v0[l]*v0[l] + v1[l]*v1[l] + v2[l]*v2[l] + v3[l]*v3[l];
    const float safe_norm = __builtin_sqrtf(s2 + 1e-7f);
    const float mlt = (s2 / (1.0f + s2)) / safe_norm;
    v0 *= mlt; v1 *= mlt; v2 *= mlt; v3 *= mlt;
    float* o = out + (size_t)t * NL;
    *(v4f*)(o + 0)  = v0;
    *(v4f*)(o + 4)  = v1;
    *(v4f*)(o + 8)  = v2;
    *(v4f*)(o + 12) = v3;
}

extern "C" void kernel_launch(void* const* d_in, const int* in_sizes, int n_in,
                              void* d_out, int out_size, void* d_ws, size_t ws_size,
                              hipStream_t stream) {
    (void)in_sizes; (void)n_in; (void)out_size; (void)ws_size;
    const float* inp = (const float*)d_in[0];   // [64, 2048, 8] fp32
    const float* W   = (const float*)d_in[1];   // [2048, 32, 8, 16] fp32
    float* out = (float*)d_out;                 // [64, 1, 32, 16, 1] fp32
    float* acc = (float*)d_ws;                  // [64, 512] fp32 accumulator

    caps_zero_kernel<<<dim3((NB * NCOL + 255) / 256), dim3(256), 0, stream>>>(acc);
    caps_gemm_kernel<<<dim3(NJ, KGROUPS), dim3(256), 0, stream>>>(inp, W, acc);
    caps_squash_kernel<<<dim3((NB * NJ + 255) / 256), dim3(256), 0, stream>>>(acc, out);
}
